// QuantizedLayer_59425167507875
// MI455X (gfx1250) — matrix operand order repaired and checked by reading, not compiled, not run on hardware
//
#include <hip/hip_runtime.h>

// ---- types for CDNA5 WMMA ----
typedef __attribute__((ext_vector_type(16))) __bf16 v16bf;
typedef __attribute__((ext_vector_type(8)))  __bf16 v8bf;
typedef __attribute__((ext_vector_type(8)))  float  v8f;
typedef __attribute__((ext_vector_type(4)))  float  v4f;

#define OUT_F 3072
#define IN_F  768
#define MROWS (8 * 4096)
#define CODE_SIZE 21

constexpr int BM = 128;    // block tile M
constexpr int BN = 256;    // block tile N
constexpr int BK = 32;     // K per WMMA step
constexpr int PITCH = 40;  // LDS row pitch in bf16 (80B: conflict-free, 16B aligned)
constexpr int ASZ = BM * PITCH;  // 5120 halfs / buffer
constexpr int BSZ = BN * PITCH;  // 10240 halfs / buffer
constexpr int KSTEPS = IN_F / BK;  // 24

// async global->LDS copy, 16B per lane; offset applies to BOTH addresses (ISA §10)
#define ASYNC_B128(ldsOff, gAddr, OFF)                                        \
  asm volatile("global_load_async_to_lds_b128 %0, %1, off offset:" #OFF       \
               :: "v"(ldsOff), "v"(gAddr) : "memory")

// ---------------------------------------------------------------------------
// Kernel 1: unpack base-8 digits from int64 codes, gather centroids -> W bf16
// W is [OUT_F, IN_F] row-major (= B^T column-major for the GEMM).
// ---------------------------------------------------------------------------
__global__ __launch_bounds__(256) void decode_weight_bf16(
    const long long* __restrict__ codes, const float* __restrict__ centroids,
    __bf16* __restrict__ wbf, int n_codes)
{
  int i = blockIdx.x * blockDim.x + threadIdx.x;
  if (i >= n_codes) return;
  unsigned long long c = (unsigned long long)codes[i];
  const long long base = (long long)i * CODE_SIZE;
  const long long n_elems = (long long)OUT_F * IN_F;
#pragma unroll
  for (int j = 0; j < CODE_SIZE; ++j) {
    float v = centroids[c & 7ull];   // centroids is [8,1]
    long long e = base + j;
    if (e < n_elems) wbf[e] = (__bf16)v;
    c >>= 3;
  }
}

// ---------------------------------------------------------------------------
// Kernel 2: y[M,N] = x[M,K] @ W[N,K]^T + bias, bf16 WMMA, f32 accumulate.
// 256 threads = 8 waves (wave32), block tile 128x256, waves 2(M) x 4(N),
// each wave owns 64x64 = 4x4 fragments of v_wmma_f32_16x16x32_bf16.
// LDS double-buffered. B tile (already bf16) is moved by the async engine
// (global_load_async_to_lds_b128, ASYNCcnt) straight into the next buffer;
// A tile is staged through VGPRs for the fp32->bf16 conversion. One
// workgroup barrier per K-step, preceded by s_wait_asynccnt 0.
// ---------------------------------------------------------------------------
__global__ __launch_bounds__(256) void qlinear_gemm_bf16(
    const float* __restrict__ x, const __bf16* __restrict__ wbf,
    const float* __restrict__ bias, float* __restrict__ y)
{
  __shared__ __bf16 As[2 * ASZ];
  __shared__ __bf16 Bs[2 * BSZ];

  const int tid  = threadIdx.x;
  const int n0   = blockIdx.x * BN;   // N fast-varying: A-block reused from L2
  const int m0   = blockIdx.y * BM;
  const int w    = tid >> 5;
  const int lane = tid & 31;
  const int waveM = w >> 2;           // 0..1 -> 64 rows
  const int waveN = w & 3;            // 0..3 -> 64 cols
  const int r  = lane & 15;           // row (A) / col (B) within fragment
  const int kh = lane >> 4;           // K-half selector

  v8f acc[4][4] = {};

  // A staging: each thread moves one contiguous 16-float half-row.
  const int lrow  = tid >> 1;         // 0..127
  const int lhalf = tid & 1;          // 0..1
  const float* aSrc = x + (long long)(m0 + lrow) * IN_F + lhalf * 16;
  // B: each thread owns one full 32-half row (64B), moved by the async engine.
  unsigned long long bGlob =
      (unsigned long long)(wbf + (long long)(n0 + tid) * IN_F);
  const unsigned bLds0 = (unsigned)(unsigned long long)(&Bs[tid * PITCH]);

  __bf16* aDst0 = &As[lrow * PITCH + lhalf * 16];

  v4f a0, a1, a2, a3;

  // ---- prologue: stage tile 0 ----
  ASYNC_B128(bLds0, bGlob, 0);
  ASYNC_B128(bLds0, bGlob, 16);
  ASYNC_B128(bLds0, bGlob, 32);
  ASYNC_B128(bLds0, bGlob, 48);
  a0 = *(const v4f*)(aSrc + 0);
  a1 = *(const v4f*)(aSrc + 4);
  a2 = *(const v4f*)(aSrc + 8);
  a3 = *(const v4f*)(aSrc + 12);
  {
    v8bf h0, h1;
#pragma unroll
    for (int t = 0; t < 4; ++t) {
      h0[t] = (__bf16)a0[t]; h0[4 + t] = (__bf16)a1[t];
      h1[t] = (__bf16)a2[t]; h1[4 + t] = (__bf16)a3[t];
    }
    *(v8bf*)(aDst0)     = h0;
    *(v8bf*)(aDst0 + 8) = h1;
  }
  asm volatile("s_wait_asynccnt 0x0" ::: "memory");
  __syncthreads();

  for (int i = 0; i < KSTEPS; ++i) {
    const int cur = i & 1;
    const bool more = (i + 1) < KSTEPS;

    // ---- issue tile i+1 loads (latency hidden by the WMMA clause) ----
    if (more) {
      // async engine writes B straight into the other LDS buffer: its last
      // readers finished before the barrier that ended step i-1.
      bGlob += BK * sizeof(__bf16);
      const unsigned bLds = bLds0 + (unsigned)((cur ^ 1) * BSZ * sizeof(__bf16));
      ASYNC_B128(bLds, bGlob, 0);
      ASYNC_B128(bLds, bGlob, 16);
      ASYNC_B128(bLds, bGlob, 32);
      ASYNC_B128(bLds, bGlob, 48);
      aSrc += BK;
      a0 = *(const v4f*)(aSrc + 0);
      a1 = *(const v4f*)(aSrc + 4);
      a2 = *(const v4f*)(aSrc + 8);
      a3 = *(const v4f*)(aSrc + 12);
      __builtin_prefetch(aSrc + BK, 0, 1);   // global_prefetch_b8: 2 steps ahead
    }

    // ---- fragments per ISA VGPR layouts ----
    // A (16-bit 16x32): lane r=M, two 8-half chunks at K = kh*8 and 16+kh*8
    const __bf16* aBuf = &As[cur * ASZ];
    const __bf16* bBuf = &Bs[cur * BSZ];
    v16bf af[4];
#pragma unroll
    for (int mi = 0; mi < 4; ++mi) {
      const __bf16* p = aBuf + (waveM * 64 + mi * 16 + r) * PITCH + kh * 8;
      v8bf lo = *(const v8bf*)(p);
      v8bf hi = *(const v8bf*)(p + 16);
      af[mi] = __builtin_shufflevector(lo, hi, 0, 1, 2, 3, 4, 5, 6, 7,
                                       8, 9, 10, 11, 12, 13, 14, 15);
    }
    // B (32x16): lane r=N-col, two 8-half chunks at K = kh*8 and 16+kh*8
    v16bf bfm[4];
#pragma unroll
    for (int ni = 0; ni < 4; ++ni) {
      const __bf16* p = bBuf + (waveN * 64 + ni * 16 + r) * PITCH + kh * 8;
      v8bf lo = *(const v8bf*)(p);
      v8bf hi = *(const v8bf*)(p + 16);
      bfm[ni] = __builtin_shufflevector(lo, hi, 0, 1, 2, 3, 4, 5, 6, 7,
                                        8, 9, 10, 11, 12, 13, 14, 15);
    }

    // ---- 16 WMMAs per wave per K-step ----
#pragma unroll
    for (int mi = 0; mi < 4; ++mi)
#pragma unroll
      for (int ni = 0; ni < 4; ++ni)
        acc[mi][ni] = __builtin_amdgcn_wmma_f32_16x16x32_bf16(
            false, af[mi], false, bfm[ni], (short)0, acc[mi][ni], false, false);

    // ---- drain staged A regs into the other LDS buffer ----
    if (more) {
      __bf16* aDst = &As[(cur ^ 1) * ASZ] + lrow * PITCH + lhalf * 16;
      v8bf h0, h1;
#pragma unroll
      for (int t = 0; t < 4; ++t) {
        h0[t] = (__bf16)a0[t]; h0[4 + t] = (__bf16)a1[t];
        h1[t] = (__bf16)a2[t]; h1[4 + t] = (__bf16)a3[t];
      }
      *(v8bf*)(aDst)     = h0;
      *(v8bf*)(aDst + 8) = h1;
    }
    asm volatile("s_wait_asynccnt 0x0" ::: "memory");
    __syncthreads();
  }

  // ---- epilogue: C/D layout -> row = kh*8 + vgpr, col = lane&15; add bias ----
#pragma unroll
  for (int mi = 0; mi < 4; ++mi)
#pragma unroll
    for (int ni = 0; ni < 4; ++ni) {
      const int col  = n0 + waveN * 64 + ni * 16 + r;
      const int rowb = m0 + waveM * 64 + mi * 16 + kh * 8;
      const float bv = bias[col];
      float* yp = y + (long long)rowb * OUT_F + col;
#pragma unroll
      for (int rr = 0; rr < 8; ++rr) {
        // streamed 403MB output: non-temporal so x/W blocks stay in L2
        __builtin_nontemporal_store(acc[mi][ni][rr] + bv,
                                    yp + (long long)rr * OUT_F);
      }
    }
}

// ---------------------------------------------------------------------------
extern "C" void kernel_launch(void* const* d_in, const int* in_sizes, int n_in,
                              void* d_out, int out_size, void* d_ws, size_t ws_size,
                              hipStream_t stream) {
  const float*     x     = (const float*)d_in[0];      // [8,4096,768] fp32
  const long long* codes = (const long long*)d_in[1];  // [112348] int64
  const float*     cent  = (const float*)d_in[2];      // [8,1] fp32
  const float*     bias  = (const float*)d_in[3];      // [3072] fp32
  float*  y   = (float*)d_out;                         // [8,4096,3072] fp32
  __bf16* wbf = (__bf16*)d_ws;                         // 3072*768*2 = 4.7MB scratch

  const int n_codes = in_sizes[1];
  decode_weight_bf16<<<(n_codes + 255) / 256, 256, 0, stream>>>(codes, cent, wbf, n_codes);

  dim3 grid(OUT_F / BN, MROWS / BM);  // (12, 256), N fast for A-block L2 reuse
  qlinear_gemm_bf16<<<grid, 256, 0, stream>>>(x, wbf, bias, y);
}
